// EDPGNN_14783277433060
// MI455X (gfx1250) — compile-verified, matrix operand order checked by reading, not run on hardware
//
#include <hip/hip_runtime.h>

typedef __attribute__((ext_vector_type(16))) _Float16 v16h;
typedef __attribute__((ext_vector_type(8)))  _Float16 h8;
typedef __attribute__((ext_vector_type(8)))  float    v8f;

static constexpr int Nn   = 1024;   // nodes
static constexpr int Fd   = 64;     // node features
static constexpr int Cc   = 4;      // edge channels in
static constexpr int Ee   = 32768;  // edges per channel
static constexpr int Hd   = 256;    // MLP hidden
static constexpr int Cout = 4;      // edge channels out

// ---------------------------------------------------------------------------
// h[n, c*F+f] = (1+eps) * x[n,f]   (broadcast self term, pre-atomic init)
// ---------------------------------------------------------------------------
__global__ void k_init_h(const float* __restrict__ x, const float* __restrict__ eps,
                         float* __restrict__ h) {
  int idx = blockIdx.x * blockDim.x + threadIdx.x;       // over N*C*F
  if (idx >= Nn * Cc * Fd) return;
  int f = idx % Fd;
  int n = idx / (Cc * Fd);
  h[idx] = (1.0f + eps[0]) * x[n * Fd + f];
}

// ---------------------------------------------------------------------------
// Per-channel edge-weighted scatter-add:  h[tgt, c*F + :] += x[src,:] * w
// ---------------------------------------------------------------------------
__global__ void k_agg(const float* __restrict__ x, const int* __restrict__ ei,
                      const float* __restrict__ ew, float* __restrict__ h) {
  int idx = blockIdx.x * blockDim.x + threadIdx.x;       // over C*E
  if (idx >= Cc * Ee) return;
  int c = idx / Ee, e = idx % Ee;
  int src = ei[(c * 2 + 0) * Ee + e];
  int tgt = ei[(c * 2 + 1) * Ee + e];
  float w = ew[c * Ee + e];
  const float* xs = x + (size_t)src * Fd;
  float* hd = h + (size_t)tgt * (Cc * Fd) + c * Fd;
#pragma unroll 4
  for (int f = 0; f < Fd; ++f) atomicAdd(&hd[f], xs[f] * w);
}

// ---------------------------------------------------------------------------
// Dense adjacency (transposed layout): adj[i, j, c] += w  for (i=src, j=tgt)
// ---------------------------------------------------------------------------
__global__ void k_adj(const int* __restrict__ ei, const float* __restrict__ ew,
                      float* __restrict__ adj) {
  int idx = blockIdx.x * blockDim.x + threadIdx.x;       // over C*E
  if (idx >= Cc * Ee) return;
  int c = idx / Ee, e = idx % Ee;
  int src = ei[(c * 2 + 0) * Ee + e];
  int tgt = ei[(c * 2 + 1) * Ee + e];
  atomicAdd(&adj[((size_t)src * Nn + tgt) * Cc + c], ew[c * Ee + e]);
}

// ---------------------------------------------------------------------------
// Pre-pack edge-MLP weights as f16:
//   W1a_h[c*H+k]  = We1[c,k]                 (adjacency rows of We1)
//   W2p[col*H+k]  = col<4 ? We2[k,col] : 0   (transposed + padded to 16 cols
//                                             -> contiguous B-fragment loads)
// ---------------------------------------------------------------------------
__global__ void k_prep(const float* __restrict__ We1, const float* __restrict__ We2,
                       _Float16* __restrict__ W1a_h, _Float16* __restrict__ W2p) {
  int t = blockIdx.x * blockDim.x + threadIdx.x;         // over 16*H
  if (t < Cc * Hd) W1a_h[t] = (_Float16)We1[t];
  if (t < 16 * Hd) {
    int col = t / Hd, k = t % Hd;
    W2p[t] = (col < Cout) ? (_Float16)We2[k * Cout + col] : (_Float16)0.0f;
  }
}

// ---------------------------------------------------------------------------
// Generic WMMA GEMM: C = act(A[M,K] @ B[K,Nc] + bias), f32 in, f16 compute,
// f32 or f16 out (compile-time). One wave per 16x16 tile.
// Wave32 fragment layout (ISA 7.12.2):
//   A: lane<16 -> row=lane,    K chunks {kb..kb+7, kb+16..kb+23}, kb=0
//      lane>=16 -> row=lane-16, kb=8
//   B: same pattern with lane -> output column
//   D: VGPR g -> row g (lanes 0-15) / row g+8 (lanes 16-31), col = lane&15
// ---------------------------------------------------------------------------
template <int RELU, int OUT_HALF>
__global__ __launch_bounds__(32)
void k_gemm(const float* __restrict__ A, const float* __restrict__ B,
            const float* __restrict__ bias, void* __restrict__ Cmat,
            int M, int K, int Nc) {
  int n0 = blockIdx.x * 16;
  int m0 = blockIdx.y * 16;
  int lane = threadIdx.x & 31;
  int r  = lane & 15;
  int kb = (lane < 16) ? 0 : 8;

  v8f acc = {};
  for (int k0 = 0; k0 < K; k0 += 32) {
    v16h af, bf;
    const float* Ap = A + (size_t)(m0 + r) * K + (k0 + kb);
    const float* Bp = B + (size_t)(k0 + kb) * Nc + (n0 + r);
#pragma unroll
    for (int t = 0; t < 8; ++t) {
      af[t]     = (_Float16)Ap[t];
      af[t + 8] = (_Float16)Ap[t + 16];
      bf[t]     = (_Float16)Bp[(size_t)t * Nc];
      bf[t + 8] = (_Float16)Bp[(size_t)(t + 16) * Nc];
    }
    acc = __builtin_amdgcn_wmma_f32_16x16x32_f16(false, af, false, bf,
                                                 (short)0, acc, false, false);
  }
  int col = n0 + r;
  int rowbase = m0 + ((lane >> 4) << 3);
  float b = bias ? bias[col] : 0.0f;                     // wave-uniform branch
#pragma unroll
  for (int g = 0; g < 8; ++g) {
    float v = acc[g] + b;
    if (RELU) v = fmaxf(v, 0.0f);
    size_t idx = (size_t)(rowbase + g) * Nc + col;
    if (OUT_HALF) ((_Float16*)Cmat)[idx] = (_Float16)v;
    else          ((float*)Cmat)[idx] = v;
  }
}

// ---------------------------------------------------------------------------
// Pair kernel: out[i,j,:] = relu(P[i] + Q[j] + adj[i,j,:]@We1[0:4]) @ We2 + be2
// Block: 256 threads (8 waves), one i per block, 128 j per block.
// The block's Q working set is the contiguous 64KB slab Qh[jb..jb+128) --
// staged into LDS with GLOBAL_LOAD_ASYNC_TO_LDS_B128 (ASYNCcnt-tracked,
// fully coalesced), XOR-swizzled in 16B chunks so the strided per-lane
// fragment reads are LDS-bank-conflict-free. Hidden build is packed f16
// (v_pk_fma/add/max), then 8 chained v_wmma_f32_16x16x32_f16.
// ---------------------------------------------------------------------------
__global__ __launch_bounds__(256)
void k_pair(const _Float16* __restrict__ Ph, const _Float16* __restrict__ Qh,
            const float* __restrict__ adj, const _Float16* __restrict__ W1a,
            const _Float16* __restrict__ W2p, const float* __restrict__ be2,
            float* __restrict__ out) {
  __shared__ _Float16 Qs[128 * Hd];            // exactly 64 KB

  int i   = blockIdx.y;
  int jb  = blockIdx.x * 128;
  int tid = threadIdx.x;

  // ---- async-stage Q[jb..jb+128) -> LDS, 16B-chunk XOR swizzle ----
  // chunk c (of 32) in row r lives at LDS offset r*512 + ((c ^ (r&31)) << 4).
  {
    uint32_t lds_base = (uint32_t)(uintptr_t)&Qs[0];     // flat->LDS offset
    uint64_t gbase = (uint64_t)(uintptr_t)(Qh + (size_t)jb * Hd);
#pragma unroll
    for (int it = 0; it < 16; ++it) {                    // 4096 chunks / 256 thr
      int g = tid + it * 256;
      int row = g >> 5, c = g & 31;
      uint32_t lds_addr = lds_base + ((uint32_t)row << 9)
                        + ((uint32_t)(c ^ (row & 31)) << 4);
      uint64_t gaddr = gbase + ((uint64_t)g << 4);
      asm volatile("global_load_async_to_lds_b128 %0, %1, off"
                   :: "v"(lds_addr), "v"(gaddr) : "memory");
    }
    asm volatile("s_wait_asynccnt 0x0" ::: "memory");
  }
  __syncthreads();

  int wave = tid >> 5;
  int lane = tid & 31;
  int r  = lane & 15;
  int kb = (lane < 16) ? 0 : 8;
  int j0 = jb + wave * 16;
  int j  = j0 + r;                     // this lane's hidden row (pair (i,j))

  const float4 a4 = *(const float4*)&adj[((size_t)i * Nn + j) * Cc];
  const _Float16 ax = (_Float16)a4.x, ay = (_Float16)a4.y;
  const _Float16 az = (_Float16)a4.z, aw = (_Float16)a4.w;

  const _Float16* Pi = Ph + (size_t)i * Hd;    // broadcast across block
  const _Float16* Bc = W2p + (size_t)r * Hd;   // per-column row of B^T
  int rrow = wave * 16 + r;                    // row inside LDS tile
  const _Float16* Qrow = Qs + (size_t)rrow * Hd;
  int m = rrow & 31;                           // swizzle key

  v8f acc = {};
#pragma unroll
  for (int k0 = 0; k0 < Hd; k0 += 32) {
    int ka = k0 + kb;
    int c0 = ka >> 3;                          // 8-half chunks
    h8 p0 = *(const h8*)(Pi + ka);             h8 p1 = *(const h8*)(Pi + ka + 16);
    h8 q0 = *(const h8*)(Qrow + ((c0 ^ m) << 3));
    h8 q1 = *(const h8*)(Qrow + (((c0 + 2) ^ m) << 3));
    h8 w00 = *(const h8*)(W1a + ka);           h8 w01 = *(const h8*)(W1a + ka + 16);
    h8 w10 = *(const h8*)(W1a + Hd + ka);      h8 w11 = *(const h8*)(W1a + Hd + ka + 16);
    h8 w20 = *(const h8*)(W1a + 2 * Hd + ka);  h8 w21 = *(const h8*)(W1a + 2 * Hd + ka + 16);
    h8 w30 = *(const h8*)(W1a + 3 * Hd + ka);  h8 w31 = *(const h8*)(W1a + 3 * Hd + ka + 16);
    h8 b0 = *(const h8*)(Bc + ka);             h8 b1 = *(const h8*)(Bc + ka + 16);

    h8 c0v = p0 + q0 + ax * w00 + ay * w10 + az * w20 + aw * w30;
    h8 c1v = p1 + q1 + ax * w01 + ay * w11 + az * w21 + aw * w31;
    h8 z = {};
    c0v = __builtin_elementwise_max(c0v, z);   // relu -> v_pk_max_num_f16
    c1v = __builtin_elementwise_max(c1v, z);

    v16h af, bf;
#pragma unroll
    for (int t = 0; t < 8; ++t) {
      af[t] = c0v[t]; af[t + 8] = c1v[t];
      bf[t] = b0[t];  bf[t + 8] = b1[t];
    }
    acc = __builtin_amdgcn_wmma_f32_16x16x32_f16(false, af, false, bf,
                                                 (short)0, acc, false, false);
  }

  if (r < Cout) {                      // only 4 of 16 output columns are real
    float bb = be2[r];
    int rowbase = j0 + ((lane >> 4) << 3);
#pragma unroll
    for (int g = 0; g < 8; ++g)
      out[((size_t)i * Nn + (rowbase + g)) * Cout + r] = acc[g] + bb;
  }
}

// ---------------------------------------------------------------------------
extern "C" void kernel_launch(void* const* d_in, const int* in_sizes, int n_in,
                              void* d_out, int out_size, void* d_ws, size_t ws_size,
                              hipStream_t stream) {
  (void)in_sizes; (void)n_in; (void)out_size; (void)ws_size;
  const float* x   = (const float*)d_in[0];
  const int*   ei  = (const int*)  d_in[1];
  const float* ew  = (const float*)d_in[2];
  const float* Wn1 = (const float*)d_in[3];
  const float* bn1 = (const float*)d_in[4];
  const float* Wn2 = (const float*)d_in[5];
  const float* bn2 = (const float*)d_in[6];
  const float* We1 = (const float*)d_in[7];
  const float* be1 = (const float*)d_in[8];
  const float* We2 = (const float*)d_in[9];
  const float* be2 = (const float*)d_in[10];
  const float* eps = (const float*)d_in[11];
  float* out = (float*)d_out;

  char* ws = (char*)d_ws;
  float*    h     = (float*)ws;                 ws += (size_t)Nn * Cc * Fd * 4;  // 1 MB
  float*    hid1  = (float*)ws;                 ws += (size_t)Nn * Hd * 4;       // 1 MB
  float*    xnext = (float*)ws;                 ws += (size_t)Nn * Fd * 4;       // 256 KB
  _Float16* Ph    = (_Float16*)ws;              ws += (size_t)Nn * Hd * 2;       // 512 KB
  _Float16* Qh    = (_Float16*)ws;              ws += (size_t)Nn * Hd * 2;       // 512 KB
  _Float16* W1a_h = (_Float16*)ws;              ws += (size_t)Cc * Hd * 2;       // 2 KB
  _Float16* W2p   = (_Float16*)ws;              ws += (size_t)16 * Hd * 2;       // 8 KB
  float*    adj   = (float*)ws;                                                  // 16 MB

  hipMemsetAsync(adj, 0, (size_t)Nn * Nn * Cc * sizeof(float), stream);

  k_init_h<<<(Nn * Cc * Fd) / 256, 256, 0, stream>>>(x, eps, h);
  k_agg   <<<(Cc * Ee) / 256, 256, 0, stream>>>(x, ei, ew, h);
  k_adj   <<<(Cc * Ee) / 256, 256, 0, stream>>>(ei, ew, adj);
  k_prep  <<<(16 * Hd) / 256, 256, 0, stream>>>(We1, We2, W1a_h, W2p);

  // Node MLP: x_next = relu(h @ Wn1 + bn1) @ Wn2 + bn2
  { dim3 g(Hd / 16, Nn / 16);
    k_gemm<1, 0><<<g, 32, 0, stream>>>(h,    Wn1, bn1, hid1,  Nn, Cc * Fd, Hd); }
  { dim3 g(Fd / 16, Nn / 16);
    k_gemm<0, 0><<<g, 32, 0, stream>>>(hid1, Wn2, bn2, xnext, Nn, Hd,      Fd); }

  // Rank decomposition of the edge MLP first layer (f16 outputs):
  //   P = x_next @ We1[C:C+F]   + be1   (x_i contribution, bias folded)
  //   Q = x_next @ We1[C+F:C+2F]        (x_j contribution)
  { dim3 g(Hd / 16, Nn / 16);
    k_gemm<0, 1><<<g, 32, 0, stream>>>(xnext, We1 + (size_t)Cc * Hd,        be1,     Ph, Nn, Fd, Hd);
    k_gemm<0, 1><<<g, 32, 0, stream>>>(xnext, We1 + (size_t)(Cc + Fd) * Hd, nullptr, Qh, Nn, Fd, Hd); }

  // Fused pair kernel: async LDS staging + packed-f16 hidden + WMMA projection
  { dim3 g(Nn / 128, Nn); k_pair<<<g, 256, 0, stream>>>(Ph, Qh, adj, W1a_h, W2p, be2, out); }
}
